// Attention_40973988003968
// MI455X (gfx1250) — matrix operand order repaired and checked by reading, not compiled, not run on hardware
//
#include <hip/hip_runtime.h>
#include <hip/hip_bf16.h>

#define DIM    1024
#define HEADS  16
#define DHEAD  64
#define SEQ    2048
#define BATCH  2
#define M_TOT  (BATCH * SEQ)    // 4096
#define QKV_N  (3 * DIM)        // 3072
#define BH     (BATCH * HEADS)  // 32
#define ASCALE 0.125f           // 64^-0.5
#define LN_EPS 1e-5f

typedef __bf16 bf16;
typedef __attribute__((ext_vector_type(16))) __bf16 v16bf;
typedef __attribute__((ext_vector_type(8)))  __bf16 bf16x8;
typedef __attribute__((ext_vector_type(8)))  float  v8f;
typedef __attribute__((ext_vector_type(4)))  float  f32x4;

// ---------------------------------------------------------------------------
// WMMA fragment helpers (CDNA5 wave32 layouts, ISA 7.12.2)
// A 16x32 bf16: lane L holds row M=L&15; lanes<16 K={0..7,16..23}, lanes>=16
// K={8..15,24..31}; element pairs contiguous in K -> two 16B loads.
// ---------------------------------------------------------------------------
static __device__ inline v16bf frag_a_bf16(const bf16* base, int ld) {
  const int lane = threadIdx.x & 31;
  const int row  = lane & 15;
  const int kb   = (lane >> 4) << 3;  // 0 or 8
  const bf16* p = base + (size_t)row * ld;
  bf16x8 lo = *(const bf16x8*)(p + kb);
  bf16x8 hi = *(const bf16x8*)(p + kb + 16);
  v16bf r;
#pragma unroll
  for (int e = 0; e < 8; ++e) { r[e] = lo[e]; r[e + 8] = hi[e]; }
  return r;
}

// B 32x16 bf16: lane L holds column N=L&15; lanes<16 K={0..7,16..23}, lanes>=16
// K={8..15,24..31}.  Sourced from B^T stored row-major (RT[n][k]) so K is contiguous.
static __device__ inline v16bf frag_b_bf16T(const bf16* baseT, int ld) {
  const int lane = threadIdx.x & 31;
  const int n  = lane & 15;
  const int ko = (lane >> 4) << 3;    // 0 or 8
  const bf16* p = baseT + (size_t)n * ld + ko;
  bf16x8 lo = *(const bf16x8*)(p);
  bf16x8 hi = *(const bf16x8*)(p + 16);
  v16bf r;
#pragma unroll
  for (int e = 0; e < 8; ++e) { r[e] = lo[e]; r[e + 8] = hi[e]; }
  return r;
}

static __device__ inline v8f wmma_bf16(v16bf a, v16bf b, v8f c) {
  return __builtin_amdgcn_wmma_f32_16x16x32_bf16(
      /*neg_a=*/false, a, /*neg_b=*/false, b,
      /*c_mod=*/(short)0, c, /*reuse_a=*/false, /*reuse_b=*/false);
}

static __device__ inline void atomicMaxF(float* addr, float v) {
  if (v >= 0.0f) atomicMax((int*)addr, __float_as_int(v));
  else           atomicMin((unsigned int*)addr, __float_as_uint(v));
}

// ---------------------------------------------------------------------------
// 0) fp32 [R][C] -> bf16 transposed [C][R]  (weights -> B^T layout)
// ---------------------------------------------------------------------------
__global__ void k_transpose_bf16(const float* __restrict__ in,
                                 bf16* __restrict__ out, int R, int C) {
  int idx = blockIdx.x * blockDim.x + threadIdx.x;
  if (idx >= R * C) return;
  int r = idx / C, c = idx - r * C;
  out[(size_t)c * R + r] = (bf16)in[idx];
}

__global__ void k_init_gmax(unsigned int* gmax_bits) {
  if (threadIdx.x == 0 && blockIdx.x == 0) *gmax_bits = 0xFF800000u;  // -inf
}

// ---------------------------------------------------------------------------
// 1) LayerNorm: one 256-thread block per row of 1024
// ---------------------------------------------------------------------------
__global__ void k_layernorm(const float* __restrict__ x,
                            const float* __restrict__ lnw,
                            const float* __restrict__ lnb,
                            bf16* __restrict__ xn) {
  __shared__ float red[256];
  const int row = blockIdx.x;
  const int t   = threadIdx.x;
  const float* xr = x + (size_t)row * DIM;
  f32x4 v = *(const f32x4*)(xr + t * 4);
  red[t] = v[0] + v[1] + v[2] + v[3];
  __syncthreads();
  for (int off = 128; off > 0; off >>= 1) {
    if (t < off) red[t] += red[t + off];
    __syncthreads();
  }
  const float mu = red[0] * (1.0f / DIM);
  __syncthreads();
  const float d0 = v[0] - mu, d1 = v[1] - mu, d2 = v[2] - mu, d3 = v[3] - mu;
  red[t] = d0 * d0 + d1 * d1 + d2 * d2 + d3 * d3;
  __syncthreads();
  for (int off = 128; off > 0; off >>= 1) {
    if (t < off) red[t] += red[t + off];
    __syncthreads();
  }
  const float rstd = rsqrtf(red[0] * (1.0f / DIM) + LN_EPS);
  f32x4 w = *(const f32x4*)(lnw + t * 4);
  f32x4 b = *(const f32x4*)(lnb + t * 4);
  bf16* o = xn + (size_t)row * DIM + t * 4;
  o[0] = (bf16)(d0 * rstd * w[0] + b[0]);
  o[1] = (bf16)(d1 * rstd * w[1] + b[1]);
  o[2] = (bf16)(d2 * rstd * w[2] + b[2]);
  o[3] = (bf16)(d3 * rstd * w[3] + b[3]);
}

// ---------------------------------------------------------------------------
// 2) QKV GEMM: xn[4096,1024] x wT^T -> scatter q(scaled)/k/vT head-major bf16.
// Block = 8 waves; block tile 128(M)x64(N); each wave 32x32 (2x2 WMMA tiles).
// ---------------------------------------------------------------------------
__global__ void k_qkv_gemm(const bf16* __restrict__ xn,
                           const bf16* __restrict__ wT,  // [3072][1024]
                           bf16* __restrict__ q, bf16* __restrict__ kbuf,
                           bf16* __restrict__ vT) {
  const int wave = threadIdx.x >> 5, lane = threadIdx.x & 31;
  const int n0 = blockIdx.x * 64 + (wave & 1) * 32;
  const int m0 = blockIdx.y * 128 + (wave >> 1) * 32;
  v8f acc[2][2] = {};
  for (int kk = 0; kk < DIM; kk += 32) {
    v16bf a0 = frag_a_bf16(xn + (size_t)m0 * DIM + kk, DIM);
    v16bf a1 = frag_a_bf16(xn + (size_t)(m0 + 16) * DIM + kk, DIM);
    v16bf b0 = frag_b_bf16T(wT + (size_t)n0 * DIM + kk, DIM);
    v16bf b1 = frag_b_bf16T(wT + (size_t)(n0 + 16) * DIM + kk, DIM);
    acc[0][0] = wmma_bf16(a0, b0, acc[0][0]);
    acc[0][1] = wmma_bf16(a0, b1, acc[0][1]);
    acc[1][0] = wmma_bf16(a1, b0, acc[1][0]);
    acc[1][1] = wmma_bf16(a1, b1, acc[1][1]);
  }
  const int col = lane & 15, rhalf = (lane >> 4) << 3;
#pragma unroll
  for (int mi = 0; mi < 2; ++mi)
#pragma unroll
    for (int ni = 0; ni < 2; ++ni)
#pragma unroll
      for (int r = 0; r < 8; ++r) {
        const int m = m0 + mi * 16 + rhalf + r;
        const int n = n0 + ni * 16 + col;
        const float val = acc[mi][ni][r];
        const int b = m >> 11, i = m & 2047;
        const int which = n >> 10, inner = n & 1023;
        const int h = inner >> 6, d = inner & 63;
        const size_t hd = (size_t)(b * HEADS + h);
        if (which == 0)       q[(hd * SEQ + i) * DHEAD + d] = (bf16)(val * ASCALE);
        else if (which == 1)  kbuf[(hd * SEQ + i) * DHEAD + d] = (bf16)val;
        else                  vT[(hd * DHEAD + d) * SEQ + i] = (bf16)val;
      }
}

// ---------------------------------------------------------------------------
// 3) sim = q @ k^T per head, write fp32 to attn region + global max reduce.
// grid (j=32, i=16, bh=32); same 128x64 block tiling as above.
// ---------------------------------------------------------------------------
__global__ void k_sim(const bf16* __restrict__ q, const bf16* __restrict__ kbuf,
                      float* __restrict__ attn, float* __restrict__ gmax) {
  __shared__ float wmax[8];
  const int wave = threadIdx.x >> 5, lane = threadIdx.x & 31;
  const int bh = blockIdx.z;
  const int j0 = blockIdx.x * 64 + (wave & 1) * 32;
  const int i0 = blockIdx.y * 128 + (wave >> 1) * 32;
  const bf16* qb = q + (size_t)bh * SEQ * DHEAD;
  const bf16* kb = kbuf + (size_t)bh * SEQ * DHEAD;
  v8f acc[2][2] = {};
#pragma unroll
  for (int d = 0; d < DHEAD; d += 32) {
    v16bf a0 = frag_a_bf16(qb + (size_t)i0 * DHEAD + d, DHEAD);
    v16bf a1 = frag_a_bf16(qb + (size_t)(i0 + 16) * DHEAD + d, DHEAD);
    v16bf b0 = frag_b_bf16T(kb + (size_t)j0 * DHEAD + d, DHEAD);
    v16bf b1 = frag_b_bf16T(kb + (size_t)(j0 + 16) * DHEAD + d, DHEAD);
    acc[0][0] = wmma_bf16(a0, b0, acc[0][0]);
    acc[0][1] = wmma_bf16(a0, b1, acc[0][1]);
    acc[1][0] = wmma_bf16(a1, b0, acc[1][0]);
    acc[1][1] = wmma_bf16(a1, b1, acc[1][1]);
  }
  float* out = attn + (size_t)bh * SEQ * SEQ;
  const int col = lane & 15, rhalf = (lane >> 4) << 3;
  float lmax = -__builtin_inff();
#pragma unroll
  for (int mi = 0; mi < 2; ++mi)
#pragma unroll
    for (int ni = 0; ni < 2; ++ni)
#pragma unroll
      for (int r = 0; r < 8; ++r) {
        const int m = i0 + mi * 16 + rhalf + r;
        const int n = j0 + ni * 16 + col;
        const float val = acc[mi][ni][r];
        out[(size_t)m * SEQ + n] = val;
        lmax = fmaxf(lmax, val);
      }
#pragma unroll
  for (int off = 16; off > 0; off >>= 1)
    lmax = fmaxf(lmax, __shfl_xor(lmax, off, 32));
  if (lane == 0) wmax[wave] = lmax;
  __syncthreads();
  if (threadIdx.x == 0) {
    float m = wmax[0];
#pragma unroll
    for (int i = 1; i < 8; ++i) m = fmaxf(m, wmax[i]);
    atomicMaxF(gmax, m);
  }
}

// ---------------------------------------------------------------------------
// 4) numerator = relu(sim - gmax) rewritten in place; rowsum; ctx = num @ v;
//    oh = ctx / (rowsum + 1e-6) in bf16 [m, h*64+d] layout.
// grid (strip=16, bh=32); 8 waves, each owns a 16-row strip x all 64 d-cols.
// ---------------------------------------------------------------------------
__global__ void k_softmax_av(float* __restrict__ attn,
                             const bf16* __restrict__ vT,
                             const float* __restrict__ gmaxp,
                             float* __restrict__ rowsum,
                             bf16* __restrict__ oh) {
  __shared__ float rs_lds[128];
  const int wave = threadIdx.x >> 5, lane = threadIdx.x & 31;
  const int bh = blockIdx.y;
  const int i0 = blockIdx.x * 128 + wave * 16;
  const int b = bh >> 4, h = bh & 15;
  const float gmax = *gmaxp;
  float* arow = attn + (size_t)bh * SEQ * SEQ;
  const bf16* vb = vT + (size_t)bh * DHEAD * SEQ;
  const int row = lane & 15;
  const int kb  = (lane >> 4) << 3;  // 0 or 8
  float* myrow = arow + (size_t)(i0 + row) * SEQ;
  v8f acc[4] = {};
  float lsum = 0.0f;
  for (int jt = 0; jt < SEQ; jt += 32) {
    f32x4 s0 = *(f32x4*)(myrow + jt + kb);
    f32x4 s1 = *(f32x4*)(myrow + jt + kb + 4);
    f32x4 s2 = *(f32x4*)(myrow + jt + kb + 16);
    f32x4 s3 = *(f32x4*)(myrow + jt + kb + 20);
    v16bf a;
#pragma unroll
    for (int e = 0; e < 4; ++e) {
      float n0 = fmaxf(s0[e] - gmax, 0.0f); lsum += n0; a[e]      = (bf16)n0; s0[e] = n0;
      float n1 = fmaxf(s1[e] - gmax, 0.0f); lsum += n1; a[e + 4]  = (bf16)n1; s1[e] = n1;
      float n2 = fmaxf(s2[e] - gmax, 0.0f); lsum += n2; a[e + 8]  = (bf16)n2; s2[e] = n2;
      float n3 = fmaxf(s3[e] - gmax, 0.0f); lsum += n3; a[e + 12] = (bf16)n3; s3[e] = n3;
    }
    *(f32x4*)(myrow + jt + kb)      = s0;
    *(f32x4*)(myrow + jt + kb + 4)  = s1;
    *(f32x4*)(myrow + jt + kb + 16) = s2;
    *(f32x4*)(myrow + jt + kb + 20) = s3;
#pragma unroll
    for (int nt = 0; nt < 4; ++nt) {
      v16bf bf = frag_b_bf16T(vb + (size_t)(nt * 16) * SEQ + jt, SEQ);
      acc[nt] = wmma_bf16(a, bf, acc[nt]);
    }
  }
  // lanes r and r+16 jointly hold row r's K-range -> pairwise exchange
  const float tot = lsum + __shfl_xor(lsum, 16, 32);
  if (lane < 16) {
    rowsum[(size_t)bh * SEQ + i0 + lane] = tot;
    rs_lds[wave * 16 + lane] = tot;
  }
  __syncthreads();
  const int col = lane & 15, rhalf = (lane >> 4) << 3;
#pragma unroll
  for (int nt = 0; nt < 4; ++nt)
#pragma unroll
    for (int r = 0; r < 8; ++r) {
      const int rl = rhalf + r;
      const float rs = rs_lds[wave * 16 + rl];
      const float val = acc[nt][r] / (rs + 1e-6f);
      const size_t m = (size_t)b * SEQ + i0 + rl;
      oh[m * DIM + h * DHEAD + nt * 16 + col] = (bf16)val;
    }
}

// ---------------------------------------------------------------------------
// 5) normalize attn region by per-row sums (vectorized grid-stride)
// ---------------------------------------------------------------------------
__global__ void k_attn_norm(float* __restrict__ attn,
                            const float* __restrict__ rowsum) {
  const size_t total4 = (size_t)BH * SEQ * SEQ / 4;
  const size_t stride = (size_t)gridDim.x * blockDim.x;
  for (size_t idx = (size_t)blockIdx.x * blockDim.x + threadIdx.x;
       idx < total4; idx += stride) {
    const size_t row = idx / (SEQ / 4);
    const float sc = 1.0f / (rowsum[row] + 1e-6f);
    f32x4 v = ((f32x4*)attn)[idx];
    v[0] *= sc; v[1] *= sc; v[2] *= sc; v[3] *= sc;
    ((f32x4*)attn)[idx] = v;
  }
}

// ---------------------------------------------------------------------------
// 6) out = oh[4096,1024] @ w_out, fp32 to d_out
// ---------------------------------------------------------------------------
__global__ void k_out_gemm(const bf16* __restrict__ oh,
                           const bf16* __restrict__ wT,  // [1024 out][1024 in]
                           float* __restrict__ out) {
  const int wave = threadIdx.x >> 5, lane = threadIdx.x & 31;
  const int n0 = blockIdx.x * 64 + (wave & 1) * 32;
  const int m0 = blockIdx.y * 128 + (wave >> 1) * 32;
  v8f acc[2][2] = {};
  for (int kk = 0; kk < DIM; kk += 32) {
    v16bf a0 = frag_a_bf16(oh + (size_t)m0 * DIM + kk, DIM);
    v16bf a1 = frag_a_bf16(oh + (size_t)(m0 + 16) * DIM + kk, DIM);
    v16bf b0 = frag_b_bf16T(wT + (size_t)n0 * DIM + kk, DIM);
    v16bf b1 = frag_b_bf16T(wT + (size_t)(n0 + 16) * DIM + kk, DIM);
    acc[0][0] = wmma_bf16(a0, b0, acc[0][0]);
    acc[0][1] = wmma_bf16(a0, b1, acc[0][1]);
    acc[1][0] = wmma_bf16(a1, b0, acc[1][0]);
    acc[1][1] = wmma_bf16(a1, b1, acc[1][1]);
  }
  const int col = lane & 15, rhalf = (lane >> 4) << 3;
#pragma unroll
  for (int mi = 0; mi < 2; ++mi)
#pragma unroll
    for (int ni = 0; ni < 2; ++ni)
#pragma unroll
      for (int r = 0; r < 8; ++r)
        out[(size_t)(m0 + mi * 16 + rhalf + r) * DIM + n0 + ni * 16 + col] =
            acc[mi][ni][r];
}

// ---------------------------------------------------------------------------
extern "C" void kernel_launch(void* const* d_in, const int* in_sizes, int n_in,
                              void* d_out, int out_size, void* d_ws,
                              size_t ws_size, hipStream_t stream) {
  (void)in_sizes; (void)n_in; (void)out_size; (void)ws_size;
  const float* x     = (const float*)d_in[0];
  const float* ln_w  = (const float*)d_in[1];
  const float* ln_b  = (const float*)d_in[2];
  const float* w_qkv = (const float*)d_in[3];
  const float* w_out = (const float*)d_in[4];

  float* out  = (float*)d_out;                          // [4096,1024]
  float* attn = out + (size_t)M_TOT * DIM;              // [32,2048,2048]

  char* w = (char*)d_ws;
  float* gmax   = (float*)w;  w += 256;
  float* rowsum = (float*)w;  w += (size_t)BH * SEQ * 4;
  bf16* xn      = (bf16*)w;   w += (size_t)M_TOT * DIM * 2;
  bf16* wqkvT   = (bf16*)w;   w += (size_t)QKV_N * DIM * 2;
  bf16* woutT   = (bf16*)w;   w += (size_t)DIM * DIM * 2;
  bf16* q       = (bf16*)w;   w += (size_t)BH * SEQ * DHEAD * 2;
  bf16* kbuf    = (bf16*)w;   w += (size_t)BH * SEQ * DHEAD * 2;
  bf16* vT      = (bf16*)w;   w += (size_t)BH * SEQ * DHEAD * 2;
  bf16* oh      = (bf16*)w;   w += (size_t)M_TOT * DIM * 2;

  k_transpose_bf16<<<(DIM * QKV_N + 255) / 256, 256, 0, stream>>>(w_qkv, wqkvT, DIM, QKV_N);
  k_transpose_bf16<<<(DIM * DIM + 255) / 256, 256, 0, stream>>>(w_out, woutT, DIM, DIM);
  k_init_gmax<<<1, 64, 0, stream>>>((unsigned int*)gmax);
  k_layernorm<<<M_TOT, 256, 0, stream>>>(x, ln_w, ln_b, xn);
  k_qkv_gemm<<<dim3(QKV_N / 64, M_TOT / 128), 256, 0, stream>>>(xn, wqkvT, q, kbuf, vT);
  k_sim<<<dim3(SEQ / 64, SEQ / 128, BH), 256, 0, stream>>>(q, kbuf, attn, gmax);
  k_softmax_av<<<dim3(SEQ / 128, BH), 256, 0, stream>>>(attn, vT, gmax, rowsum, oh);
  k_attn_norm<<<8192, 256, 0, stream>>>(attn, rowsum);
  k_out_gemm<<<dim3(DIM / 64, M_TOT / 128), 256, 0, stream>>>(oh, woutT, out);
}
